// SwinTransformerBlock_12764642804222
// MI455X (gfx1250) — matrix operand order repaired and checked by reading, not compiled, not run on hardware
//
#include <hip/hip_runtime.h>
#include <hip/hip_bf16.h>

// ---------------------------------------------------------------------------
// Swin Transformer block for gfx1250 (MI455X), bf16 WMMA everywhere.
// Two fused kernels: (attn: LN1+shift+window+QKV+MSA+proj+residual) and
// (mlp: LN2+fc1+GELU+fc2+residual).  Weights pre-converted to bf16 in d_ws.
// Round 3: async global->LDS staging with correct builtin pointer types.
// ---------------------------------------------------------------------------

typedef __attribute__((ext_vector_type(16))) __bf16 v16bf;
typedef __attribute__((ext_vector_type(8)))  __bf16 v8bf;
typedef __attribute__((ext_vector_type(8)))  float  v8f;
typedef __attribute__((ext_vector_type(4)))  int    v4i;

#define BB     64
#define HH     56
#define WI     56
#define CC     96
#define HEADS  3
#define HD     32
#define WIN    7
#define SHFT   3
#define WWT    49       // real tokens per window
#define TPW    64       // padded tokens per window
#define MLPH   384

#if defined(__gfx1250__) && \
    __has_builtin(__builtin_amdgcn_global_load_async_to_lds_b128) && \
    __has_builtin(__builtin_amdgcn_s_wait_asynccnt)
#define USE_ASYNC_LDS 1
typedef __attribute__((address_space(1))) v4i gv4i;   // global int4
typedef __attribute__((address_space(3))) v4i lv4i;   // LDS int4
#define ASYNC_CP16(dst, src) \
  __builtin_amdgcn_global_load_async_to_lds_b128((gv4i*)(src), (lv4i*)(dst), 0, 0)
#else
#define USE_ASYNC_LDS 0
#endif

// A-fragment: 16x32 bf16 tile, rows across lanes (ISA 7.12.2 16-bit A layout).
__device__ __forceinline__ v16bf load_a16(const __bf16* tile_row0, int ld,
                                          int k0, int lane) {
  const int m  = lane & 15;
  const int ko = (lane >> 4) * 8;
  const __bf16* p = tile_row0 + m * ld + k0 + ko;
  v8bf lo = *(const v8bf*)(p);
  v8bf hi = *(const v8bf*)(p + 16);
  v16bf r;
#pragma unroll
  for (int i = 0; i < 8; ++i) { r[i] = lo[i]; r[i + 8] = hi[i]; }
  return r;
}

// B-fragment: 32x16 bf16 tile with B[k,n] = src[(n0+n)*ld + k0 + k]
__device__ __forceinline__ v16bf load_bt16(const __bf16* src, int ld,
                                           int n0, int k0, int lane) {
  const int n  = lane & 15;
  const int ko = (lane >> 4) * 8;
  const __bf16* p = src + (n0 + n) * ld + k0 + ko;
  v8bf lo = *(const v8bf*)(p);
  v8bf hi = *(const v8bf*)(p + 16);
  v16bf r;
#pragma unroll
  for (int i = 0; i < 8; ++i) { r[i] = lo[i]; r[i + 8] = hi[i]; }
  return r;
}

__device__ __forceinline__ v8f wmma_bf16(v16bf a, v16bf b, v8f c) {
  return __builtin_amdgcn_wmma_f32_16x16x32_bf16(
      false, a, false, b, (short)0, c, false, false);
}

// ---------------------------------------------------------------------------
__global__ void swin_cvt_bf16(const float* __restrict__ s,
                              __bf16* __restrict__ d, int n) {
  int i = blockIdx.x * 256 + threadIdx.x;
  if (i < n) d[i] = (__bf16)s[i];
}

// ---------------------------------------------------------------------------
// Kernel 1: one block per window (4096 blocks x 128 threads = 4 waves).
// LDS arena (phased, overlapped):
//   [0,12288)      XS  bf16 [64][96]  -> later PS bf16 [64][64]
//   [12288,36864)  XF  f32  [64][96]  -> later QS,KS -> later YS f32
//   [36864,49152)  VT  bf16 [96][64]
//   [49152,61440)  OS  bf16 [64][96]
// ---------------------------------------------------------------------------
__global__ __launch_bounds__(128) void swin_attn_kernel(
    const float* __restrict__ x,
    const float* __restrict__ n1w, const float* __restrict__ n1b,
    const __bf16* __restrict__ qkvw,   // [288][96] bf16
    const float* __restrict__ qkvb,
    const float* __restrict__ relb,    // [169][3]
    const __bf16* __restrict__ projw,  // [96][96] bf16
    const float* __restrict__ projb,
    float* __restrict__ x1)            // == d_out
{
  __shared__ char smem[61440];
  __shared__ int  rowOff[WWT];
  __shared__ int  regid[WWT];

  __bf16* XS = (__bf16*)(smem);
  __bf16* PS = (__bf16*)(smem);
  float*  XF = (float*)(smem + 12288);
  __bf16* QS = (__bf16*)(smem + 12288);
  __bf16* KS = (__bf16*)(smem + 24576);
  __bf16* VT = (__bf16*)(smem + 36864);
  __bf16* OS = (__bf16*)(smem + 49152);
  float*  YS = (float*)(smem + 12288);

  const int tid  = threadIdx.x;
  const int lane = tid & 31;
  const int wv   = tid >> 5;
  const int m0   = wv * 16;
  const int win  = blockIdx.x;
  const int b    = win >> 6;
  const int wi   = win & 63;
  const int wh   = wi >> 3;
  const int wc   = wi & 7;

  if (tid < 108) __builtin_prefetch(qkvw + tid * 256, 0, 1);

  // per-token metadata: global row offset (shift folded in) + mask region id
  if (tid < WWT) {
    int r = tid / WIN, c = tid - (tid / WIN) * WIN;
    int gr = wh * WIN + r, gc = wc * WIN + c;
    int h0 = gr + SHFT; if (h0 >= HH) h0 -= HH;
    int w0 = gc + SHFT; if (w0 >= WI) w0 -= WI;
    rowOff[tid] = (b * (HH * WI) + h0 * WI + w0) * CC;
    int rr = (gr < HH - WIN) ? 0 : ((gr < HH - SHFT) ? 1 : 2);
    int rc = (gc < WI - WIN) ? 0 : ((gc < WI - SHFT) ? 1 : 2);
    regid[tid] = rr * 3 + rc;
  }
  __syncthreads();

  // stage token rows (fp32) into XF; zero-pad XS rows 49..63
#if USE_ASYNC_LDS
  for (int idx = tid; idx < WWT * (CC / 4); idx += 128) {   // 16B chunks
    int t = idx / (CC / 4), part = (idx - t * (CC / 4)) * 4;
    ASYNC_CP16(XF + t * CC + part, x + rowOff[t] + part);
  }
#else
  for (int idx = tid; idx < WWT * CC; idx += 128) {
    int t = idx / CC, ch = idx - t * CC;
    XF[t * CC + ch] = x[rowOff[t] + ch];
  }
#endif
  for (int idx = tid; idx < (TPW - WWT) * CC; idx += 128)
    XS[WWT * CC + idx] = (__bf16)0.0f;
#if USE_ASYNC_LDS
  __builtin_amdgcn_s_wait_asynccnt(0);
#endif
  __syncthreads();

  // LayerNorm1 -> XS (bf16)
  if (tid < WWT) {
    float s = 0.f, s2 = 0.f;
    for (int c = 0; c < CC; ++c) { float v = XF[tid * CC + c]; s += v; s2 += v * v; }
    float mu   = s * (1.0f / CC);
    float var  = s2 * (1.0f / CC) - mu * mu;
    float rstd = rsqrtf(var + 1e-5f);
    for (int c = 0; c < CC; ++c) {
      float v = (XF[tid * CC + c] - mu) * rstd * n1w[c] + n1b[c];
      XS[tid * CC + c] = (__bf16)v;
    }
  }
  __syncthreads();

  const int n  = lane & 15;
  const int hi = lane >> 4;
  const float qscale = 0.17677669529663687f;   // 1/sqrt(32)

  // ---- QKV GEMM: A-fragments loaded once, reused over 18 N-tiles
  {
    v16bf aq[3];
#pragma unroll
    for (int ks = 0; ks < 3; ++ks)
      aq[ks] = load_a16(XS + m0 * CC, CC, ks * 32, lane);

    for (int nt = 0; nt < 18; ++nt) {
      v8f acc = {};
#pragma unroll
      for (int ks = 0; ks < 3; ++ks)
        acc = wmma_bf16(aq[ks], load_bt16(qkvw, CC, nt * 16, ks * 32, lane), acc);

      int   o    = nt * 16 + n;
      float bias = qkvb[o];
      if (nt < 6) {                    // Q (pre-scaled)
#pragma unroll
        for (int v = 0; v < 8; ++v)
          QS[(m0 + v + 8 * hi) * CC + o] = (__bf16)((acc[v] + bias) * qscale);
      } else if (nt < 12) {            // K
        int f = o - 96;
#pragma unroll
        for (int v = 0; v < 8; ++v)
          KS[(m0 + v + 8 * hi) * CC + f] = (__bf16)(acc[v] + bias);
      } else {                         // V, feature-major for the AV B-frag
        int f = o - 192;
#pragma unroll
        for (int v = 0; v < 8; ++v)
          VT[f * TPW + (m0 + v + 8 * hi)] = (__bf16)(acc[v] + bias);
      }
    }
  }
  __syncthreads();

  // ---- per-head attention
  for (int h = 0; h < HEADS; ++h) {
    v8f sc[4];
    {
      v16bf aQ = load_a16(QS + m0 * CC + h * HD, CC, 0, lane);  // hoisted
#pragma unroll
      for (int nt = 0; nt < 4; ++nt) {
        v8f z = {};
        sc[nt] = wmma_bf16(aQ, load_bt16(KS + h * HD, CC, nt * 16, 0, lane), z);
      }
    }
#pragma unroll
    for (int v = 0; v < 8; ++v) {
      int i = m0 + v + 8 * hi;       // query token
      float vals[4];
#pragma unroll
      for (int nt = 0; nt < 4; ++nt) {
        int j = nt * 16 + n;         // key token
        float s = sc[nt][v];
        if (j < WWT && i < WWT) {
          int ri = i / WIN, ci = i - ri * WIN;
          int rj = j / WIN, cj = j - rj * WIN;
          int ridx = (ri - rj + WIN - 1) * (2 * WIN - 1) + (ci - cj + WIN - 1);
          s += relb[ridx * HEADS + h];
          if (regid[i] != regid[j]) s -= 100.0f;
        }
        if (j >= WWT) s = -1e30f;
        vals[nt] = s;
      }
      float mx = fmaxf(fmaxf(vals[0], vals[1]), fmaxf(vals[2], vals[3]));
      for (int msk = 8; msk >= 1; msk >>= 1) mx = fmaxf(mx, __shfl_xor(mx, msk, 32));
      float sum = 0.f;
#pragma unroll
      for (int nt = 0; nt < 4; ++nt) { vals[nt] = __expf(vals[nt] - mx); sum += vals[nt]; }
      for (int msk = 8; msk >= 1; msk >>= 1) sum += __shfl_xor(sum, msk, 32);
      float inv = 1.0f / sum;
#pragma unroll
      for (int nt = 0; nt < 4; ++nt)
        PS[i * TPW + nt * 16 + n] = (__bf16)(vals[nt] * inv);
    }
    __syncthreads();

    // AV: [16,64] x [64,32] -> OS columns h*32..h*32+31
    {
      v16bf aP[2];
#pragma unroll
      for (int ks = 0; ks < 2; ++ks)
        aP[ks] = load_a16(PS + m0 * TPW, TPW, ks * 32, lane);
#pragma unroll
      for (int nt = 0; nt < 2; ++nt) {
        v8f acc = {};
#pragma unroll
        for (int ks = 0; ks < 2; ++ks)
          acc = wmma_bf16(aP[ks],
                          load_bt16(VT + h * HD * TPW, TPW, nt * 16, ks * 32, lane),
                          acc);
        int f = h * HD + nt * 16 + n;
#pragma unroll
        for (int v = 0; v < 8; ++v)
          OS[(m0 + v + 8 * hi) * CC + f] = (__bf16)acc[v];
      }
    }
    __syncthreads();
  }

  // ---- output projection + bias -> YS (f32), then residual + scatter-back
  {
    v16bf ao[3];
#pragma unroll
    for (int ks = 0; ks < 3; ++ks)
      ao[ks] = load_a16(OS + m0 * CC, CC, ks * 32, lane);
#pragma unroll
    for (int nt = 0; nt < 6; ++nt) {
      v8f acc = {};
#pragma unroll
      for (int ks = 0; ks < 3; ++ks)
        acc = wmma_bf16(ao[ks], load_bt16(projw, CC, nt * 16, ks * 32, lane), acc);
      int   f    = nt * 16 + n;
      float bias = projb[f];
#pragma unroll
      for (int v = 0; v < 8; ++v)
        YS[(m0 + v + 8 * hi) * CC + f] = acc[v] + bias;
    }
  }
  __syncthreads();

  for (int idx = tid; idx < WWT * CC; idx += 128) {
    int t = idx / CC, ch = idx - t * CC;
    x1[rowOff[t] + ch] = YS[t * CC + ch] + x[rowOff[t] + ch];
  }
}

// ---------------------------------------------------------------------------
// Kernel 2: MLP over 64-token tiles (3136 blocks x 128 threads).
// LDS arena:
//   [0,12288)      X2S bf16 [64][96]
//   [12288,36864)  XF  f32 [64][96] -> later low half of HS, then YS
//   [12288,61440)  HS  bf16 [64][384]
// ---------------------------------------------------------------------------
__global__ __launch_bounds__(128) void swin_mlp_kernel(
    float* __restrict__ xio,           // d_out: in = x1, out = final
    const float* __restrict__ n2w, const float* __restrict__ n2b,
    const __bf16* __restrict__ fc1w,   // [384][96] bf16
    const float* __restrict__ fc1b,
    const __bf16* __restrict__ fc2w,   // [96][384] bf16
    const float* __restrict__ fc2b)
{
  __shared__ char smem[61440];
  __bf16* X2S = (__bf16*)(smem);
  float*  XF  = (float*)(smem + 12288);
  __bf16* HS  = (__bf16*)(smem + 12288);
  float*  YS  = (float*)(smem + 12288);

  const int tid  = threadIdx.x;
  const int lane = tid & 31;
  const int wv   = tid >> 5;
  const int m0   = wv * 16;
  const int n    = lane & 15;
  const int hi   = lane >> 4;
  float* xt = xio + (long)blockIdx.x * 64 * CC;

  if (tid < 96) __builtin_prefetch(fc1w + tid * 384, 0, 1);

#if USE_ASYNC_LDS
  for (int idx = tid; idx < 64 * (CC / 4); idx += 128)      // 16B chunks
    ASYNC_CP16(XF + idx * 4, xt + idx * 4);
  __builtin_amdgcn_s_wait_asynccnt(0);
#else
  for (int idx = tid; idx < 64 * CC; idx += 128) XF[idx] = xt[idx];
#endif
  __syncthreads();

  if (tid < 64) {
    float s = 0.f, s2 = 0.f;
    for (int c = 0; c < CC; ++c) { float v = XF[tid * CC + c]; s += v; s2 += v * v; }
    float mu = s * (1.0f / CC), var = s2 * (1.0f / CC) - mu * mu;
    float rstd = rsqrtf(var + 1e-5f);
    for (int c = 0; c < CC; ++c)
      X2S[tid * CC + c] = (__bf16)((XF[tid * CC + c] - mu) * rstd * n2w[c] + n2b[c]);
  }
  __syncthreads();

  // fc1 + exact GELU -> HS (A-fragments hoisted over the 24 N-tiles)
  {
    v16bf ax[3];
#pragma unroll
    for (int ks = 0; ks < 3; ++ks)
      ax[ks] = load_a16(X2S + m0 * CC, CC, ks * 32, lane);

    for (int nt = 0; nt < 24; ++nt) {
      v8f acc = {};
#pragma unroll
      for (int ks = 0; ks < 3; ++ks)
        acc = wmma_bf16(ax[ks], load_bt16(fc1w, CC, nt * 16, ks * 32, lane), acc);
      int   f    = nt * 16 + n;
      float bias = fc1b[f];
#pragma unroll
      for (int v = 0; v < 8; ++v) {
        float g = acc[v] + bias;
        g = 0.5f * g * (1.0f + erff(g * 0.7071067811865475f));
        HS[(m0 + v + 8 * hi) * MLPH + f] = (__bf16)g;
      }
    }
  }
  __syncthreads();

  // fc2: K = 384, k-outer so each A-fragment is loaded once; 6 resident accs
  v8f acc2[6];
  {
    v8f z = {};
#pragma unroll
    for (int nt = 0; nt < 6; ++nt) acc2[nt] = z;
  }
  for (int ks = 0; ks < 12; ++ks) {
    v16bf a = load_a16(HS + m0 * MLPH, MLPH, ks * 32, lane);
#pragma unroll
    for (int nt = 0; nt < 6; ++nt)
      acc2[nt] = wmma_bf16(a, load_bt16(fc2w, MLPH, nt * 16, ks * 32, lane), acc2[nt]);
  }
  __syncthreads();   // everyone done with HS before YS overwrites it

#pragma unroll
  for (int nt = 0; nt < 6; ++nt) {
    int   f    = nt * 16 + n;
    float bias = fc2b[f];
#pragma unroll
    for (int v = 0; v < 8; ++v)
      YS[(m0 + v + 8 * hi) * CC + f] = acc2[nt][v] + bias;
  }
  __syncthreads();

  for (int idx = tid; idx < 64 * CC; idx += 128)
    xt[idx] = YS[idx] + xt[idx];     // residual (read-before-write, same thread)
}

// ---------------------------------------------------------------------------
extern "C" void kernel_launch(void* const* d_in, const int* in_sizes, int n_in,
                              void* d_out, int out_size, void* d_ws, size_t ws_size,
                              hipStream_t stream) {
  const float* x     = (const float*)d_in[0];
  const float* n1w   = (const float*)d_in[1];
  const float* n1b   = (const float*)d_in[2];
  const float* qkv_w = (const float*)d_in[3];
  const float* qkv_b = (const float*)d_in[4];
  const float* relb  = (const float*)d_in[5];
  const float* projw = (const float*)d_in[6];
  const float* projb = (const float*)d_in[7];
  const float* n2w   = (const float*)d_in[8];
  const float* n2b   = (const float*)d_in[9];
  const float* fc1w  = (const float*)d_in[10];
  const float* fc1b  = (const float*)d_in[11];
  const float* fc2w  = (const float*)d_in[12];
  const float* fc2b  = (const float*)d_in[13];
  float* out = (float*)d_out;

  __bf16* wb      = (__bf16*)d_ws;
  __bf16* qkvw_b  = wb;                 // 27648 elems
  __bf16* projw_b = wb + 27648;         //  9216
  __bf16* fc1w_b  = wb + 36864;         // 36864
  __bf16* fc2w_b  = wb + 73728;         // 36864

  swin_cvt_bf16<<<(27648 + 255) / 256, 256, 0, stream>>>(qkv_w, qkvw_b, 27648);
  swin_cvt_bf16<<<(9216  + 255) / 256, 256, 0, stream>>>(projw, projw_b, 9216);
  swin_cvt_bf16<<<(36864 + 255) / 256, 256, 0, stream>>>(fc1w,  fc1w_b, 36864);
  swin_cvt_bf16<<<(36864 + 255) / 256, 256, 0, stream>>>(fc2w,  fc2w_b, 36864);

  swin_attn_kernel<<<4096, 128, 0, stream>>>(x, n1w, n1b, qkvw_b, qkv_b, relb,
                                             projw_b, projb, out);
  swin_mlp_kernel<<<3136, 128, 0, stream>>>(out, n2w, n2b, fc1w_b, fc1b,
                                            fc2w_b, fc2b);
}
